// CustomTransformerEncoderOnly_23029614641426
// MI455X (gfx1250) — matrix operand order repaired and checked by reading, not compiled, not run on hardware
//
#include <hip/hip_runtime.h>

typedef __attribute__((ext_vector_type(16))) _Float16 v16h;
typedef __attribute__((ext_vector_type(8)))  float    v8f;
typedef __attribute__((ext_vector_type(4)))  int      v4i;

constexpr int B_ = 8, T_ = 512, E_ = 1024, H_ = 16, L_ = 4, FF_ = 4096;
constexpr int D_ = E_ / H_;           // 64
constexpr int M_ = B_ * T_;           // 4096 rows of activations

#if defined(__AMDGCN__) && __has_builtin(__builtin_amdgcn_global_load_async_to_lds_b128) && __has_builtin(__builtin_amdgcn_s_wait_asynccnt)
#define USE_ASYNC_LDS 1
#define AS1 __attribute__((address_space(1)))
#define AS3 __attribute__((address_space(3)))
#else
#define USE_ASYNC_LDS 0
#endif

__device__ inline void wait_async_lds() {
#if USE_ASYNC_LDS
  __builtin_amdgcn_s_wait_asynccnt(0);
#endif
}

union frag_u { v16h h; uint4 u[2]; };

// A-frag (16x32 f16, lane=M row, regs=K): lane L row=L&15;
// halves j=0..7 -> k=(L>>4)*8 + j ; j=8..15 -> k=16+(L>>4)*8+(j-8)
__device__ inline v16h frag_a_lds(const _Float16* base, int stride) {
  const int lane = threadIdx.x & 31;
  const int r = lane & 15, hi = lane >> 4;
  const _Float16* p = base + r * stride + hi * 8;
  frag_u f;
  f.u[0] = *(const uint4*)(p);
  f.u[1] = *(const uint4*)(p + 16);
  return f.h;
}

// B-frag (32x16 f16, lane=N col, regs=K): lane L col=L&15;
// halves j=0..7 -> k=(L>>4)*8 + j ; j=8..15 -> k=16+(L>>4)*8+(j-8)
__device__ inline v16h frag_b_lds(const _Float16* base, int stride) {
  const int lane = threadIdx.x & 31;
  const int r = lane & 15, hi = lane >> 4;
  const _Float16* p = base + r * stride + hi * 8;
  frag_u f;
  f.u[0] = *(const uint4*)(p);
  f.u[1] = *(const uint4*)(p + 16);
  return f.h;
}

__device__ inline v8f zero_v8f() {
  v8f z = {0.f, 0.f, 0.f, 0.f, 0.f, 0.f, 0.f, 0.f};
  return z;
}

// 16-byte global -> LDS transfer: async DMA on CDNA5, sync fallback otherwise
__device__ inline void xfer_b128(const _Float16* g, _Float16* s) {
#if USE_ASYNC_LDS
  __builtin_amdgcn_global_load_async_to_lds_b128((AS1 v4i*)(uintptr_t)g, (AS3 v4i*)(uintptr_t)s, 0, 0);
#else
  *(uint4*)s = *(const uint4*)g;
#endif
}

// ---------------------------------------------------------------------------
// Generic f32 -> f16 conversion (weights), grid-stride
// ---------------------------------------------------------------------------
__global__ void cvt_f32_f16(const float* __restrict__ in, _Float16* __restrict__ out, size_t n) {
  size_t i = (size_t)blockIdx.x * blockDim.x + threadIdx.x;
  size_t stride = (size_t)gridDim.x * blockDim.x;
  for (; i < n; i += stride) out[i] = (_Float16)in[i];
}

// ---------------------------------------------------------------------------
// Embedding: x = src + pe(b)  (pe indexes BATCH per the reference)
// ---------------------------------------------------------------------------
__global__ __launch_bounds__(256) void embed_kernel(const float* __restrict__ src,
                                                    float* __restrict__ x32,
                                                    _Float16* __restrict__ x16) {
  const int row = blockIdx.x;            // b*T + t
  const int b = row / T_;
  const size_t base = (size_t)row * E_;
  for (int c = threadIdx.x; c < E_; c += 256) {
    int half2i = (c >> 1) * 2;
    float div = __expf((float)half2i * (-9.210340371976184f / (float)E_));
    float a = (float)b * div;
    float pe = (c & 1) ? __cosf(a) : __sinf(a);
    float v = src[base + c] + pe;
    x32[base + c] = v;
    x16[base + c] = (_Float16)v;
  }
}

// ---------------------------------------------------------------------------
// WMMA GEMM: out[M,N] = A[M,K](f16) @ W[N,K](f16)^T + bias
// MODE 0: f32 out, 1: f16 out, 2: relu + f16 out
// block = 256 threads (8 waves), tile 128x128, K-step 32,
// double-buffered LDS with async global->LDS prefetch of the next K-tile.
// ---------------------------------------------------------------------------
constexpr int TM = 128, TN = 128, TK = 32, LPAD = 40;

template <int MODE>
__global__ __launch_bounds__(256) void gemm_f16(const _Float16* __restrict__ A,
                                                const _Float16* __restrict__ W,
                                                const float* __restrict__ bias,
                                                float* __restrict__ out32,
                                                _Float16* __restrict__ out16,
                                                int M, int N, int K) {
  __shared__ _Float16 As[2][TM * LPAD];
  __shared__ _Float16 Bs[2][TN * LPAD];
  const int tid = threadIdx.x;
  const int lane = tid & 31;
  const int wave = tid >> 5;          // 0..7
  const int wm = wave >> 2;           // 0..1 -> 64-row half
  const int wn = wave & 3;            // 0..3 -> 32-col slice
  const int mbase = blockIdx.y * TM;
  const int nbase = blockIdx.x * TN;
  const int lo = lane & 15, hi = lane >> 4;

  v8f acc[4][2];
#pragma unroll
  for (int i = 0; i < 4; ++i)
#pragma unroll
    for (int j = 0; j < 2; ++j) acc[i][j] = zero_v8f();

  // stage one 128x32 tile of A and W into LDS buffer `buf`
  auto stage = [&](int buf, int k0) {
#pragma unroll
    for (int i = 0; i < 2; ++i) {
      int c = tid * 2 + i;                 // 0..511
      int row = c >> 2, col = (c & 3) * 8;
      xfer_b128(A + (size_t)(mbase + row) * K + k0 + col, &As[buf][row * LPAD + col]);
      xfer_b128(W + (size_t)(nbase + row) * K + k0 + col, &Bs[buf][row * LPAD + col]);
    }
  };

  const int nt = K / TK;
  stage(0, 0);
  wait_async_lds();
  __syncthreads();

  for (int t = 0; t < nt; ++t) {
    const int cur = t & 1;
    if (t + 1 < nt) stage(cur ^ 1, (t + 1) * TK);   // async prefetch next tile

    v16h bf0 = frag_b_lds(&Bs[cur][(wn * 32 + 0) * LPAD], LPAD);
    v16h bf1 = frag_b_lds(&Bs[cur][(wn * 32 + 16) * LPAD], LPAD);
#pragma unroll
    for (int i = 0; i < 4; ++i) {
      v16h af = frag_a_lds(&As[cur][(wm * 64 + i * 16) * LPAD], LPAD);
      acc[i][0] = __builtin_amdgcn_wmma_f32_16x16x32_f16(false, af, false, bf0,
                                                         (short)0, acc[i][0], false, false);
      acc[i][1] = __builtin_amdgcn_wmma_f32_16x16x32_f16(false, af, false, bf1,
                                                         (short)0, acc[i][1], false, false);
    }
    wait_async_lds();
    __syncthreads();
  }

  // epilogue: C layout row = e + 8*hi, col = lo
  const int cn0 = nbase + wn * 32;
#pragma unroll
  for (int i = 0; i < 4; ++i) {
    int rbase = mbase + wm * 64 + i * 16 + hi * 8;
#pragma unroll
    for (int j = 0; j < 2; ++j) {
      int col = cn0 + j * 16 + lo;
      float bv = bias ? bias[col] : 0.f;
#pragma unroll
      for (int e = 0; e < 8; ++e) {
        float v = acc[i][j][e] + bv;
        size_t idx = (size_t)(rbase + e) * N + col;
        if (MODE == 0) {
          out32[idx] = v;
        } else {
          if (MODE == 2) v = v > 0.f ? v : 0.f;
          out16[idx] = (_Float16)v;
        }
      }
    }
  }
}

// ---------------------------------------------------------------------------
// Fused attention: one block per (b*H+h, q-tile of 64). 128 threads = 4 waves,
// each wave owns 16 q rows. Online softmax tracking Sum(e) and Sum(e*g) so the
// final scale 1/(Sum(e*g) + 1e-5*Sum(e)) matches softmax*gb/(sum+1e-5).
// ---------------------------------------------------------------------------
__global__ __launch_bounds__(128) void attn_kernel(const _Float16* __restrict__ qkv,
                                                   _Float16* __restrict__ o16) {
  const int bh = blockIdx.x;
  const int b = bh / H_, h = bh % H_;
  const int qbase = blockIdx.y * 64;
  const int tid = threadIdx.x;
  const int lane = tid & 31, wave = tid >> 5;
  const int lo = lane & 15, hi = lane >> 4;

  __shared__ _Float16 Qs[64 * 72];
  __shared__ _Float16 Ks[32 * 72];
  __shared__ _Float16 Vt[64 * 40];     // transposed: Vt[d][k], 64 x 32 (+pad)
  __shared__ _Float16 Ps[4][16 * 40];

  const size_t rstride = 3 * E_;
  const _Float16* qptr = qkv + (size_t)b * T_ * rstride + (size_t)h * D_;
  const _Float16* kptr = qptr + E_;
  const _Float16* vptr = qptr + 2 * E_;

  // stage Q tile (64 x 64) via async DMA
#pragma unroll
  for (int i = 0; i < 4; ++i) {
    int c = tid + i * 128;               // 0..511
    int row = c >> 3, col = (c & 7) * 8;
    xfer_b128(qptr + (size_t)(qbase + row) * rstride + col, &Qs[row * 72 + col]);
  }
  wait_async_lds();
  __syncthreads();
  const v16h qa0 = frag_a_lds(&Qs[(wave * 16) * 72], 72);       // d 0..31
  const v16h qa1 = frag_a_lds(&Qs[(wave * 16) * 72 + 32], 72);  // d 32..63

  v8f o[4];
  float mrun[8], lp[8], lg[8];
#pragma unroll
  for (int d = 0; d < 4; ++d) o[d] = zero_v8f();
#pragma unroll
  for (int e = 0; e < 8; ++e) { mrun[e] = -1e30f; lp[e] = 0.f; lg[e] = 0.f; }

  const float rsqD = 0.125f;                 // 1/sqrt(64)
  const float gcoef = -1.0f / 32768.0f;      // -1/(2*(T/4)^2)

  for (int kt = 0; kt < T_ / 32; ++kt) {
    const int tb = kt * 32;
    __syncthreads();
    // stage K (32x64, async) and V transposed (64x32, scalar scatter)
#pragma unroll
    for (int i = 0; i < 2; ++i) {
      int c = tid + i * 128;               // 0..255
      int row = c >> 3, col = (c & 7) * 8;
      xfer_b128(kptr + (size_t)(tb + row) * rstride + col, &Ks[row * 72 + col]);
      uint4 vv = *(const uint4*)(vptr + (size_t)(tb + row) * rstride + col);
      const _Float16* hv = (const _Float16*)&vv;
#pragma unroll
      for (int u = 0; u < 8; ++u) Vt[(col + u) * 40 + row] = hv[u];
    }
    wait_async_lds();
    __syncthreads();

    // scores S = Q @ K^T  (two 16x16 tiles: cols tb..tb+15, tb+16..tb+31)
    v8f s[2];
    s[0] = zero_v8f(); s[1] = zero_v8f();
    {
      v16h kb = frag_b_lds(&Ks[0], 72);
      s[0] = __builtin_amdgcn_wmma_f32_16x16x32_f16(false, qa0, false, kb, (short)0, s[0], false, false);
      kb = frag_b_lds(&Ks[0] + 32, 72);
      s[0] = __builtin_amdgcn_wmma_f32_16x16x32_f16(false, qa1, false, kb, (short)0, s[0], false, false);
      kb = frag_b_lds(&Ks[16 * 72], 72);
      s[1] = __builtin_amdgcn_wmma_f32_16x16x32_f16(false, qa0, false, kb, (short)0, s[1], false, false);
      kb = frag_b_lds(&Ks[16 * 72] + 32, 72);
      s[1] = __builtin_amdgcn_wmma_f32_16x16x32_f16(false, qa1, false, kb, (short)0, s[1], false, false);
    }

    // online softmax update (rows: e + 8*hi, cols: tb + j*16 + lo)
    _Float16* myP = &Ps[wave][0];
    float sc[8];
#pragma unroll
    for (int e = 0; e < 8; ++e) {
      const int qrow = qbase + wave * 16 + e + 8 * hi;
      float sv[2], lgb[2];
#pragma unroll
      for (int j = 0; j < 2; ++j) {
        int kcol = tb + j * 16 + lo;
        float dd = (float)(qrow - kcol);
        lgb[j] = dd * dd * gcoef;
        sv[j] = s[j][e] * rsqD;
      }
      float rm = fmaxf(sv[0], sv[1]);
#pragma unroll
      for (int off = 1; off < 16; off <<= 1) rm = fmaxf(rm, __shfl_xor(rm, off, 32));
      float mo = mrun[e];
      float mn = fmaxf(mo, rm);
      sc[e] = __expf(mo - mn);
      float se = 0.f, seg = 0.f;
#pragma unroll
      for (int j = 0; j < 2; ++j) {
        float ex = __expf(sv[j] - mn);
        float pv = ex * __expf(lgb[j]);
        myP[(e + 8 * hi) * 40 + j * 16 + lo] = (_Float16)pv;
        se += ex; seg += pv;
      }
#pragma unroll
      for (int off = 1; off < 16; off <<= 1) {
        se += __shfl_xor(se, off, 32);
        seg += __shfl_xor(seg, off, 32);
      }
      lp[e] = lp[e] * sc[e] + se;
      lg[e] = lg[e] * sc[e] + seg;
      mrun[e] = mn;
    }

    // rescale O and accumulate P @ V
#pragma unroll
    for (int d = 0; d < 4; ++d)
#pragma unroll
      for (int e = 0; e < 8; ++e) o[d][e] *= sc[e];

    const v16h pa = frag_a_lds(&Ps[wave][0], 40);   // P is 16 x 32 (K = key idx)
#pragma unroll
    for (int d = 0; d < 4; ++d) {
      v16h vb = frag_b_lds(&Vt[(d * 16) * 40], 40); // lane=d col, regs=k (contig in Vt)
      o[d] = __builtin_amdgcn_wmma_f32_16x16x32_f16(false, pa, false, vb, (short)0, o[d], false, false);
    }
  }

  // final normalization and store: o16[b, t, h*64 + d]
  _Float16* optr = o16 + ((size_t)b * T_ + qbase + wave * 16) * E_ + (size_t)h * D_;
  float inv[8];
#pragma unroll
  for (int e = 0; e < 8; ++e) inv[e] = 1.0f / (lg[e] + 1e-5f * lp[e]);
#pragma unroll
  for (int d = 0; d < 4; ++d)
#pragma unroll
    for (int e = 0; e < 8; ++e)
      optr[(size_t)(e + 8 * hi) * E_ + d * 16 + lo] = (_Float16)(o[d][e] * inv[e]);
}

// ---------------------------------------------------------------------------
// Residual + LayerNorm (in-place on x32), also refresh x16
// ---------------------------------------------------------------------------
__global__ __launch_bounds__(256) void residual_ln(const float* __restrict__ x,
                                                   const float* __restrict__ y,
                                                   const float* __restrict__ g,
                                                   const float* __restrict__ bta,
                                                   float* __restrict__ xout,
                                                   _Float16* __restrict__ x16) {
  const int row = blockIdx.x;
  const int tid = threadIdx.x;
  const size_t base = (size_t)row * E_;
  __shared__ float red[256];

  float v[4];
  float s = 0.f;
#pragma unroll
  for (int i = 0; i < 4; ++i) {
    int c = tid + i * 256;
    v[i] = x[base + c] + y[base + c];
    s += v[i];
  }
  red[tid] = s;
  __syncthreads();
  for (int w = 128; w > 0; w >>= 1) {
    if (tid < w) red[tid] += red[tid + w];
    __syncthreads();
  }
  const float mu = red[0] * (1.0f / (float)E_);
  __syncthreads();

  float s2 = 0.f;
#pragma unroll
  for (int i = 0; i < 4; ++i) {
    float d = v[i] - mu;
    s2 += d * d;
  }
  red[tid] = s2;
  __syncthreads();
  for (int w = 128; w > 0; w >>= 1) {
    if (tid < w) red[tid] += red[tid + w];
    __syncthreads();
  }
  const float rinv = rsqrtf(red[0] * (1.0f / (float)E_) + 1e-5f);

#pragma unroll
  for (int i = 0; i < 4; ++i) {
    int c = tid + i * 256;
    float o = (v[i] - mu) * rinv * g[c] + bta[c];
    xout[base + c] = o;
    x16[base + c] = (_Float16)o;
  }
}

// ---------------------------------------------------------------------------
// Head: out[b] = dot(x[b, T-1, :], head_w) + head_b
// ---------------------------------------------------------------------------
__global__ __launch_bounds__(256) void head_kernel(const float* __restrict__ x32,
                                                   const float* __restrict__ hw,
                                                   const float* __restrict__ hb,
                                                   float* __restrict__ out) {
  const int b = blockIdx.x;
  const int tid = threadIdx.x;
  const float* xr = x32 + ((size_t)b * T_ + (T_ - 1)) * E_;
  __shared__ float red[256];
  float s = 0.f;
  for (int c = tid; c < E_; c += 256) s += xr[c] * hw[c];
  red[tid] = s;
  __syncthreads();
  for (int w = 128; w > 0; w >>= 1) {
    if (tid < w) red[tid] += red[tid + w];
    __syncthreads();
  }
  if (tid == 0) out[b] = red[0] + hb[0];
}

// ---------------------------------------------------------------------------
extern "C" void kernel_launch(void* const* d_in, const int* in_sizes, int n_in,
                              void* d_out, int out_size, void* d_ws, size_t ws_size,
                              hipStream_t stream) {
  const float* src       = (const float*)d_in[0];
  const float* in_proj_w = (const float*)d_in[1];
  const float* in_proj_b = (const float*)d_in[2];
  const float* out_w     = (const float*)d_in[3];
  const float* out_b     = (const float*)d_in[4];
  const float* ff1_w     = (const float*)d_in[5];
  const float* ff1_b     = (const float*)d_in[6];
  const float* ff2_w     = (const float*)d_in[7];
  const float* ff2_b     = (const float*)d_in[8];
  const float* ln1_g     = (const float*)d_in[9];
  const float* ln1_b     = (const float*)d_in[10];
  const float* ln2_g     = (const float*)d_in[11];
  const float* ln2_b     = (const float*)d_in[12];
  const float* head_w    = (const float*)d_in[13];
  const float* head_b    = (const float*)d_in[14];

  char* ws = (char*)d_ws;
  size_t off = 0;
  auto alloc = [&](size_t bytes) -> char* {
    char* p = ws + off;
    off += (bytes + 255) & ~(size_t)255;
    return p;
  };
  _Float16* wqkv16 = (_Float16*)alloc((size_t)L_ * 3 * E_ * E_ * 2);
  _Float16* wout16 = (_Float16*)alloc((size_t)L_ * E_ * E_ * 2);
  _Float16* wff1_16 = (_Float16*)alloc((size_t)L_ * FF_ * E_ * 2);
  _Float16* wff2_16 = (_Float16*)alloc((size_t)L_ * E_ * FF_ * 2);
  float*    x32   = (float*)alloc((size_t)M_ * E_ * 4);
  _Float16* x16   = (_Float16*)alloc((size_t)M_ * E_ * 2);
  _Float16* qkv16 = (_Float16*)alloc((size_t)M_ * 3 * E_ * 2);
  _Float16* o16   = (_Float16*)alloc((size_t)M_ * E_ * 2);
  float*    y32   = (float*)alloc((size_t)M_ * E_ * 4);
  _Float16* m16   = (_Float16*)alloc((size_t)M_ * FF_ * 2);

  // convert all weights to f16 (every call; deterministic)
  {
    size_t n1 = (size_t)L_ * 3 * E_ * E_;
    cvt_f32_f16<<<(unsigned)((n1 + 255) / 256), 256, 0, stream>>>(in_proj_w, wqkv16, n1);
    size_t n2 = (size_t)L_ * E_ * E_;
    cvt_f32_f16<<<(unsigned)((n2 + 255) / 256), 256, 0, stream>>>(out_w, wout16, n2);
    size_t n3 = (size_t)L_ * FF_ * E_;
    cvt_f32_f16<<<(unsigned)((n3 + 255) / 256), 256, 0, stream>>>(ff1_w, wff1_16, n3);
    size_t n4 = (size_t)L_ * E_ * FF_;
    cvt_f32_f16<<<(unsigned)((n4 + 255) / 256), 256, 0, stream>>>(ff2_w, wff2_16, n4);
  }

  embed_kernel<<<M_, 256, 0, stream>>>(src, x32, x16);

  for (int l = 0; l < L_; ++l) {
    // QKV projection: [4096,3072] = x16 @ Wqkv^T
    gemm_f16<1><<<dim3(3 * E_ / TN, M_ / TM), 256, 0, stream>>>(
        x16, wqkv16 + (size_t)l * 3 * E_ * E_, in_proj_b + (size_t)l * 3 * E_,
        nullptr, qkv16, M_, 3 * E_, E_);

    // fused attention
    attn_kernel<<<dim3(B_ * H_, T_ / 64), 128, 0, stream>>>(qkv16, o16);

    // output projection -> y32
    gemm_f16<0><<<dim3(E_ / TN, M_ / TM), 256, 0, stream>>>(
        o16, wout16 + (size_t)l * E_ * E_, out_b + (size_t)l * E_,
        y32, nullptr, M_, E_, E_);

    residual_ln<<<M_, 256, 0, stream>>>(x32, y32, ln1_g + (size_t)l * E_,
                                        ln1_b + (size_t)l * E_, x32, x16);

    // FF1 (relu, f16 out)
    gemm_f16<2><<<dim3(FF_ / TN, M_ / TM), 256, 0, stream>>>(
        x16, wff1_16 + (size_t)l * FF_ * E_, ff1_b + (size_t)l * FF_,
        nullptr, m16, M_, FF_, E_);

    // FF2 -> y32
    gemm_f16<0><<<dim3(E_ / TN, M_ / TM), 256, 0, stream>>>(
        m16, wff2_16 + (size_t)l * E_ * FF_, ff2_b + (size_t)l * E_,
        y32, nullptr, M_, E_, FF_);

    residual_ln<<<M_, 256, 0, stream>>>(x32, y32, ln2_g + (size_t)l * E_,
                                        ln2_b + (size_t)l * E_, x32, x16);
  }

  head_kernel<<<B_, 256, 0, stream>>>(x32, head_w, head_b, (float*)d_out);
}
